// TensorSquare_59914793779561
// MI455X (gfx1250) — matrix operand order repaired and checked by reading, not compiled, not run on hardware
//
#include <hip/hip_runtime.h>
#include <hip/hip_bf16.h>
#include <stdint.h>

// ---------------------------------------------------------------------------
// TensorSquare on MI455X (gfx1250): bf16 WMMA GEMMs with on-the-fly feature
// (A-matrix) generation and pre-scaled, N-major bf16 weights (B-matrix).
//
// v2 changes vs v1:
//   * M register-blocking: each wave owns TWO 16-row M-tiles, so every B
//     fragment feeds 2 WMMAs (halves L2 B-traffic, denser XDL issue).
//   * float->bf16 via __bf16 cast so clang emits hardware packed cvt
//     instead of the integer RNE sequence (cuts feature-gen VALU).
//   * out1: u-index is uniform per 32-wide K-chunk -> hoist s[row][u].
//
// Workspace layout (bytes):
//   W0t : 64 x 4160 bf16  (N-major)   @ 0         (532,480 B)
//   W3t : 64 x 4096 bf16  (N-major)   @ 532480    (524,288 B)
//   W6t : 64 x 2016 bf16  (N-major)   @ 1056768   (258,048 B)
//   uv  : 2016 u32 (u | v<<6)         @ 1314816   (8,064 B)
// ---------------------------------------------------------------------------

typedef __attribute__((ext_vector_type(16))) __bf16 v16bf;
typedef __attribute__((ext_vector_type(8)))  float  v8f;

union Frag32 {
  uint4          q[2];
  v16bf          h;
  __bf16         bh[16];
  unsigned short us[16];
};

__device__ __forceinline__ unsigned short f2bf_bits(float f) {
  unsigned int u = __float_as_uint(f);
  u += 0x7fffu + ((u >> 16) & 1u);          // RNE (prep kernel only)
  return (unsigned short)(u >> 16);
}

__device__ __forceinline__ v8f wmma_bf16(v16bf a, v16bf b, v8f c) {
  // D = A(16x32 bf16) * B(32x16 bf16) + C(16x16 f32)
  return __builtin_amdgcn_wmma_f32_16x16x32_bf16(
      /*neg_a=*/false, a, /*neg_b=*/false, b,
      /*c_mod=*/(short)0, c, /*reuse_a=*/false, /*reuse_b=*/false);
}

// B fragment: 32x16 bf16 from N-major weights. lane holds col = lane&15,
// K = kbase..kbase+7 and kbase+16..kbase+23 (kbase includes +8 for lanes 16-31).
__device__ __forceinline__ v16bf loadB(const unsigned short* __restrict__ Bt,
                                       int ldk, int col, int kbase) {
  const uint4* p = (const uint4*)(Bt + (size_t)col * ldk + kbase);
  Frag32 f;
  f.q[0] = p[0];
  f.q[1] = p[2];
  return f.h;
}

// ---------------------------------------------------------------------------
// prep: scale + bf16-convert + transpose weights, build triu pair table
// ---------------------------------------------------------------------------
__global__ void ts_prep(const float* __restrict__ w1, const float* __restrict__ w2,
                        const float* __restrict__ w3, const float* __restrict__ w4,
                        const float* __restrict__ w5, const float* __restrict__ w6,
                        unsigned short* __restrict__ W0t,
                        unsigned short* __restrict__ W3t,
                        unsigned short* __restrict__ W6t,
                        unsigned int*   __restrict__ uvtab) {
  const int i = blockIdx.x * blockDim.x + threadIdx.x;

  const float S1 = rsqrtf(4160.0f);    // C1
  const float S2 = rsqrtf(12480.0f);   // C2 = sqrt(1/(3*4160))
  const float S4 = rsqrtf(12480.0f);   // C4/sqrt(3)
  const float S5 = rsqrtf(62400.0f);   // C5/sqrt(3) = sqrt(1/(15*4160))
  const float S3 = 1.0f / 64.0f;       // C3/sqrt(3) = sqrt(1/4096)
  const float S6 = rsqrtf(4032.0f);    // C6/sqrt(6) = sqrt(1/4032)

  const int N0 = 64 * 4160;
  const int N3 = 64 * 4096;
  const int N6 = 64 * 2016;

  if (i < N0) {
    int n = i / 4160, k = i % 4160;
    float v;
    if (k < 2016)      v = S1 * w1[(size_t)k * 64 + n];
    else if (k < 2080) v = S2 * w2[(size_t)(k - 2016) * 64 + n];
    else if (k < 4096) v = S4 * w4[(size_t)(k - 2080) * 64 + n];
    else               v = S5 * w5[(size_t)(k - 4096) * 64 + n];
    W0t[i] = f2bf_bits(v);
  } else if (i < N0 + N3) {
    int j = i - N0;
    int n = j >> 12, k = j & 4095;                  // k = u*64+v
    W3t[j] = f2bf_bits(S3 * w3[(size_t)k * 64 + n]);
  } else if (i < N0 + N3 + N6) {
    int j = i - N0 - N3;
    int n = j / 2016, k = j % 2016;
    W6t[j] = f2bf_bits(S6 * w6[(size_t)k * 64 + n]);
  } else if (i < N0 + N3 + N6 + 2016) {
    int q = i - (N0 + N3 + N6);
    int u = 0, rem = q;
    while (rem >= 63 - u) { rem -= 63 - u; ++u; }   // triu_indices order
    int v = u + 1 + rem;
    uvtab[q] = (unsigned int)(u | (v << 6));
  }
}

// ---------------------------------------------------------------------------
// main: grid = (Z/64, 7). 64 threads = 2 waves; each wave owns TWO 16-row
// M-tiles (32 rows) and all 4 N-tiles of its output group. blockIdx.y:
//   0    -> out0 (K = 4160 stacked features)
//   1..3 -> out1, k = y-1  (K = 4096, A = s_u * v_{v,k})
//   4..6 -> out2, k = y-4  (K = 2016, A = cross_k(v_u, v_v))
// ---------------------------------------------------------------------------
#define ZT   64   // rows per block (2 waves x 2 M-tiles x 16)
#define PADR 65   // LDS row stride: (r*65+u)%64 distinct across lanes

__global__ __launch_bounds__(64) void ts_main(
    const float* __restrict__ x,
    const unsigned short* __restrict__ W0t,
    const unsigned short* __restrict__ W3t,
    const unsigned short* __restrict__ W6t,
    const unsigned int*   __restrict__ uvtab,
    float* __restrict__ out) {
  __shared__ float        sS[ZT][PADR];
  __shared__ float        sV[3][ZT][PADR];
  __shared__ unsigned int sUV[2016];

  const int tid = threadIdx.x;
  const float* xt = x + (size_t)blockIdx.x * ZT * 256;

  // Stage x tile: s -> sS, interleaved 3-vectors -> sV[k]
  for (int i = tid; i < ZT * 256; i += 64) {
    int r = i >> 8, c = i & 255;
    float val = xt[i];
    if (c < 64) sS[r][c] = val;
    else { int cc = c - 64; sV[cc % 3][r][cc / 3] = val; }
  }
  for (int q = tid; q < 2016; q += 64) sUV[q] = uvtab[q];
  __syncthreads();

  const int lane = tid & 31;
  const int wv   = tid >> 5;
  const int r0   = wv * 32 + (lane & 15);        // A row, M-tile 0
  const int r1   = r0 + 16;                      // A row, M-tile 1
  const int khi  = (lane & 16) ? 8 : 0;          // A K sub-offset (ISA layout)
  const int bcol = lane & 15;                    // B column within N-tile
  const int blo  = (lane & 16) ? 8 : 0;          // B K sub-offset
  const int rowo = (lane & 16) ? 8 : 0;          // C/D row offset
  const int zr   = blockIdx.x * ZT + wv * 32;    // global row of M-tile 0
  const int y    = blockIdx.y;

  const v8f zero = {0.f, 0.f, 0.f, 0.f, 0.f, 0.f, 0.f, 0.f};
  v8f acc0[4] = {zero, zero, zero, zero};
  v8f acc1[4] = {zero, zero, zero, zero};

  if (y == 0) {
    // -------- out0 : four stacked feature segments vs W0t (ldk=4160) -------
    auto pairS = [&](int r, int c0) -> v16bf {
      Frag32 a;
#pragma unroll
      for (int j = 0; j < 16; ++j) {
        unsigned int uv = sUV[c0 + khi + j + (j & 8)];
        int u = uv & 63, v = (uv >> 6) & 63;
        a.bh[j] = (__bf16)(sS[r][u] * sS[r][v]);
      }
      return a.h;
    };
    auto diagS = [&](int r, int c0) -> v16bf {
      Frag32 a;
#pragma unroll
      for (int j = 0; j < 16; ++j) {
        float s = sS[r][c0 + khi + j + (j & 8)];
        a.bh[j] = (__bf16)(s * s);
      }
      return a.h;
    };
    auto pairV = [&](int r, int c0) -> v16bf {
      Frag32 a;
#pragma unroll
      for (int j = 0; j < 16; ++j) {
        unsigned int uv = sUV[c0 + khi + j + (j & 8)];
        int u = uv & 63, v = (uv >> 6) & 63;
        float d = sV[0][r][u] * sV[0][r][v]
                + sV[1][r][u] * sV[1][r][v]
                + sV[2][r][u] * sV[2][r][v];
        a.bh[j] = (__bf16)d;
      }
      return a.h;
    };
    auto diagV = [&](int r, int c0) -> v16bf {
      Frag32 a;
#pragma unroll
      for (int j = 0; j < 16; ++j) {
        int u = c0 + khi + j + (j & 8);
        float d = sV[0][r][u] * sV[0][r][u]
                + sV[1][r][u] * sV[1][r][u]
                + sV[2][r][u] * sV[2][r][u];
        a.bh[j] = (__bf16)d;
      }
      return a.h;
    };

    for (int c0 = 0; c0 < 2016; c0 += 32) {
      v16bf a0 = pairS(r0, c0), a1 = pairS(r1, c0);
#pragma unroll
      for (int n = 0; n < 4; ++n) {
        v16bf b = loadB(W0t, 4160, 16 * n + bcol, c0 + blo);
        acc0[n] = wmma_bf16(a0, b, acc0[n]);
        acc1[n] = wmma_bf16(a1, b, acc1[n]);
      }
    }
    for (int c0 = 0; c0 < 64; c0 += 32) {
      v16bf a0 = diagS(r0, c0), a1 = diagS(r1, c0);
#pragma unroll
      for (int n = 0; n < 4; ++n) {
        v16bf b = loadB(W0t, 4160, 16 * n + bcol, 2016 + c0 + blo);
        acc0[n] = wmma_bf16(a0, b, acc0[n]);
        acc1[n] = wmma_bf16(a1, b, acc1[n]);
      }
    }
    for (int c0 = 0; c0 < 2016; c0 += 32) {
      v16bf a0 = pairV(r0, c0), a1 = pairV(r1, c0);
#pragma unroll
      for (int n = 0; n < 4; ++n) {
        v16bf b = loadB(W0t, 4160, 16 * n + bcol, 2080 + c0 + blo);
        acc0[n] = wmma_bf16(a0, b, acc0[n]);
        acc1[n] = wmma_bf16(a1, b, acc1[n]);
      }
    }
    for (int c0 = 0; c0 < 64; c0 += 32) {
      v16bf a0 = diagV(r0, c0), a1 = diagV(r1, c0);
#pragma unroll
      for (int n = 0; n < 4; ++n) {
        v16bf b = loadB(W0t, 4160, 16 * n + bcol, 4096 + c0 + blo);
        acc0[n] = wmma_bf16(a0, b, acc0[n]);
        acc1[n] = wmma_bf16(a1, b, acc1[n]);
      }
    }
#pragma unroll
    for (int j = 0; j < 8; ++j) {
      float* op0 = out + (size_t)(zr + rowo + j) * 448;
      float* op1 = op0 + (size_t)16 * 448;
#pragma unroll
      for (int n = 0; n < 4; ++n) {
        op0[16 * n + bcol] = acc0[n][j];
        op1[16 * n + bcol] = acc1[n][j];
      }
    }
  } else if (y <= 3) {
    // -------- out1 (k = y-1): A[u*64+v] = s_u * v_{v,k} vs W3t (ldk=4096) --
    const int k = y - 1;
    for (int c0 = 0; c0 < 4096; c0 += 32) {
      const int uu    = c0 >> 6;        // uniform across the 32-wide K-chunk
      const int vbase = (c0 & 63) + khi;
      const float su0 = sS[r0][uu];
      const float su1 = sS[r1][uu];
      Frag32 a0, a1;
#pragma unroll
      for (int j = 0; j < 16; ++j) {
        int vv = vbase + j + (j & 8);
        a0.bh[j] = (__bf16)(su0 * sV[k][r0][vv]);
        a1.bh[j] = (__bf16)(su1 * sV[k][r1][vv]);
      }
      if ((c0 & 511) == 0) {
        // keep the big w3 stream warm (global_prefetch_b8)
        __builtin_prefetch(W3t + (size_t)bcol * 4096 + c0 + 1024, 0, 1);
      }
#pragma unroll
      for (int n = 0; n < 4; ++n) {
        v16bf b = loadB(W3t, 4096, 16 * n + bcol, c0 + blo);
        acc0[n] = wmma_bf16(a0.h, b, acc0[n]);
        acc1[n] = wmma_bf16(a1.h, b, acc1[n]);
      }
    }
#pragma unroll
    for (int j = 0; j < 8; ++j) {
      float* op0 = out + (size_t)(zr + rowo + j) * 448 + 64 + k;
      float* op1 = op0 + (size_t)16 * 448;
#pragma unroll
      for (int n = 0; n < 4; ++n) {
        op0[3 * (16 * n + bcol)] = acc0[n][j];
        op1[3 * (16 * n + bcol)] = acc1[n][j];
      }
    }
  } else {
    // -------- out2 (k = y-4): A[q] = cross_k(v_u, v_v) vs W6t (ldk=2016) ---
    const int k  = y - 4;
    const int k1 = (k + 1) % 3, k2 = (k + 2) % 3;
    for (int c0 = 0; c0 < 2016; c0 += 32) {
      Frag32 a0, a1;
#pragma unroll
      for (int j = 0; j < 16; ++j) {
        unsigned int uv = sUV[c0 + khi + j + (j & 8)];
        int u = uv & 63, v = (uv >> 6) & 63;
        a0.bh[j] = (__bf16)(sV[k1][r0][u] * sV[k2][r0][v]
                          - sV[k2][r0][u] * sV[k1][r0][v]);
        a1.bh[j] = (__bf16)(sV[k1][r1][u] * sV[k2][r1][v]
                          - sV[k2][r1][u] * sV[k1][r1][v]);
      }
#pragma unroll
      for (int n = 0; n < 4; ++n) {
        v16bf b = loadB(W6t, 2016, 16 * n + bcol, c0 + blo);
        acc0[n] = wmma_bf16(a0.h, b, acc0[n]);
        acc1[n] = wmma_bf16(a1.h, b, acc1[n]);
      }
    }
#pragma unroll
    for (int j = 0; j < 8; ++j) {
      float* op0 = out + (size_t)(zr + rowo + j) * 448 + 256 + k;
      float* op1 = op0 + (size_t)16 * 448;
#pragma unroll
      for (int n = 0; n < 4; ++n) {
        op0[3 * (16 * n + bcol)] = acc0[n][j];
        op1[3 * (16 * n + bcol)] = acc1[n][j];
      }
    }
  }
}

// ---------------------------------------------------------------------------
extern "C" void kernel_launch(void* const* d_in, const int* in_sizes, int n_in,
                              void* d_out, int out_size, void* d_ws, size_t ws_size,
                              hipStream_t stream) {
  const float* x  = (const float*)d_in[0];
  const float* w1 = (const float*)d_in[1];
  const float* w2 = (const float*)d_in[2];
  const float* w3 = (const float*)d_in[3];
  const float* w4 = (const float*)d_in[4];
  const float* w5 = (const float*)d_in[5];
  const float* w6 = (const float*)d_in[6];
  float* out = (float*)d_out;

  unsigned short* W0t = (unsigned short*)d_ws;
  unsigned short* W3t = W0t + 64 * 4160;
  unsigned short* W6t = W3t + 64 * 4096;
  unsigned int*   uvt = (unsigned int*)(W6t + 64 * 2016);  // 4-byte aligned

  const int total = 64 * 4160 + 64 * 4096 + 64 * 2016 + 2016;
  ts_prep<<<(total + 255) / 256, 256, 0, stream>>>(w1, w2, w3, w4, w5, w6,
                                                   W0t, W3t, W6t, uvt);

  dim3 grid(8192 / ZT, 7);
  ts_main<<<grid, 64, 0, stream>>>(x, W0t, W3t, W6t, uvt, out);
}
